// MyAttention_17033840296422
// MI455X (gfx1250) — matrix operand order repaired and checked by reading, not compiled, not run on hardware
//
#include <hip/hip_runtime.h>
#include <hip/hip_bf16.h>

#define SEQ   1024
#define BSZ   2
#define DIM   512
#define NQ    4
#define NK    4
#define NPROJ (DIM * NQ)   // 2048
#define MROWS (BSZ * SEQ)  // 2048

typedef __bf16 bf16_t;
typedef __bf16 v16bf __attribute__((ext_vector_type(16)));
typedef __bf16 v8bf  __attribute__((ext_vector_type(8)));
typedef float  v8f   __attribute__((ext_vector_type(8)));

// ---------------------------------------------------------------------------
// WMMA fragment loaders (bf16 16x16x32, wave32; A/B share a contiguous k order, not the ISA table order)
// A 16x32 (MxK): lane L holds row M=L&15; 16 contiguous K at half*16
// where half = L>>4.
// B 32x16 (KxN) loaded from B^T row-major [N][K]: lane L holds col N=L&15,
// 16 contiguous K at half*16.
// C/D 16x16 f32: vgpr r, lane L -> (m = r + (L>=16)*8, n = L&15).
// ---------------------------------------------------------------------------
static __device__ __forceinline__ v16bf load_a_frag(const bf16_t* __restrict__ base,
                                                    int lda, int row, int half, int kbase) {
  union { v16bf v; v8bf h[2]; } u;
  const bf16_t* p = base + (size_t)row * lda + kbase + half * 16;
  u.h[0] = *(const v8bf*)(p);
  u.h[1] = *(const v8bf*)(p + 8);
  return u.v;
}

static __device__ __forceinline__ v16bf load_b_frag(const bf16_t* __restrict__ baseT,
                                                    int ldb, int col, int half, int kbase) {
  return *(const v16bf*)(baseT + (size_t)col * ldb + kbase + half * 16);
}

static __device__ __forceinline__ v8f wmma_bf16(v16bf a, v16bf b, v8f c) {
  return __builtin_amdgcn_wmma_f32_16x16x32_bf16(false, a, false, b, (short)0, c, false, false);
}

// ---------------------------------------------------------------------------
// fp32 -> bf16 conversion kernels
// ---------------------------------------------------------------------------
__global__ void cvt_f32_bf16(const float* __restrict__ src, bf16_t* __restrict__ dst, int n) {
  int idx = blockIdx.x * blockDim.x + threadIdx.x;
  if (idx < n) dst[idx] = (bf16_t)src[idx];
}

// x: (seq, bs, dim) f32  ->  xt: (bs, seq, dim) bf16
__global__ void cvt_x_kernel(const float* __restrict__ x, bf16_t* __restrict__ xt, int n) {
  int idx = blockIdx.x * blockDim.x + threadIdx.x;
  if (idx >= n) return;
  int d = idx % DIM;
  int b = (idx / DIM) % BSZ;
  int s = idx / (DIM * BSZ);
  xt[((size_t)(b * SEQ + s)) * DIM + d] = (bf16_t)x[idx];
}

// ---------------------------------------------------------------------------
// Generic bf16 WMMA GEMM: C[M][N] = (A[M][K] @ Bt[N][K]^T + bias) * scale
// Block = 256 threads = 8 waves; wave w owns a 16(M) x 64(N) strip.
// mode 0: store bf16 row-major [M][N]
// mode 1: store bf16 as Vt[b][i][d][s]   (m=(b,s), n=(i,d))
// mode 2: store f32 as out[s][b][n]      (final layout (seq, bs, dim))
// ---------------------------------------------------------------------------
__global__ __launch_bounds__(256) void gemm_bf16_kernel(
    const bf16_t* __restrict__ A, const bf16_t* __restrict__ Bt,
    const float* __restrict__ bias, void* __restrict__ C,
    int M, int N, int K, float scale, int mode) {
  const int lane  = threadIdx.x & 31;
  const int wave  = threadIdx.x >> 5;
  const int half  = lane >> 4;
  const int nlane = lane & 15;
  const int m0 = blockIdx.x * 16;
  const int n0 = blockIdx.y * 512 + wave * 64;
  if (m0 >= M || n0 >= N) return;

  v8f acc[4];
#pragma unroll
  for (int j = 0; j < 4; ++j) acc[j] = (v8f)(0.0f);

  const bf16_t* Abase = A + (size_t)m0 * K;
  for (int kb = 0; kb < K; kb += 32) {
    v16bf af = load_a_frag(Abase, K, nlane, half, kb);
#pragma unroll
    for (int j = 0; j < 4; ++j) {
      v16bf bfr = load_b_frag(Bt, K, n0 + 16 * j + nlane, half, kb);
      acc[j] = wmma_bf16(af, bfr, acc[j]);
    }
  }

#pragma unroll
  for (int j = 0; j < 4; ++j) {
    const int n  = n0 + 16 * j + nlane;
    const float bv = bias[n];
#pragma unroll
    for (int r = 0; r < 8; ++r) {
      const int m = m0 + r + half * 8;
      const float v = (acc[j][r] + bv) * scale;
      if (mode == 0) {
        ((bf16_t*)C)[(size_t)m * N + n] = (bf16_t)v;
      } else if (mode == 1) {
        const int b = m >> 10, s = m & (SEQ - 1);
        const int i = n >> 9,  d = n & (DIM - 1);
        ((bf16_t*)C)[(((size_t)(b * NK + i) * DIM + d) * SEQ) + s] = (bf16_t)v;
      } else {
        const int b = m >> 10, s = m & (SEQ - 1);
        ((float*)C)[(size_t)s * (BSZ * DIM) + (size_t)b * DIM + n] = v;
      }
    }
  }
}

// ---------------------------------------------------------------------------
// Relu-attention core. Grid: (SEQ/16, NQ, BSZ), block = 256 threads (8 waves).
// Each workgroup: 16 queries x full causal key range over all NK heads.
// Wave w: score sub-tile for keys [t0+16w, t0+16w+16) then PV for d-slice
// [64w, 64w+64). Unnormalized accumulation; single divide at the end.
// ---------------------------------------------------------------------------
__global__ __launch_bounds__(256) void attn_kernel(
    const bf16_t* __restrict__ Qm,   // [MROWS][NPROJ], pre-scaled by 1/sqrt(DIM)
    const bf16_t* __restrict__ Km,   // [MROWS][NPROJ]
    const bf16_t* __restrict__ Vt,   // [BSZ][NK][DIM][SEQ]
    bf16_t* __restrict__ Amid) {     // [MROWS][NPROJ]
  __shared__ bf16_t P[16 * 128];
  __shared__ float denom[16];

  const int lane  = threadIdx.x & 31;
  const int wave  = threadIdx.x >> 5;
  const int half  = lane >> 4;
  const int nlane = lane & 15;
  const int s0 = blockIdx.x * 16;
  const int h  = blockIdx.y;
  const int b  = blockIdx.z;
  const int smax = s0 + 15;

  if (threadIdx.x < 16) denom[threadIdx.x] = 0.0f;

  // Q fragments register-resident: 16 k-steps of 32 cover DIM=512
  const bf16_t* qbase = Qm + ((size_t)(b * SEQ + s0)) * NPROJ + h * DIM;
  v16bf qf[16];
#pragma unroll
  for (int kk = 0; kk < 16; ++kk)
    qf[kk] = load_a_frag(qbase, NPROJ, nlane, half, kk * 32);

  v8f acc[4];
#pragma unroll
  for (int j = 0; j < 4; ++j) acc[j] = (v8f)(0.0f);
  float dsum[8] = {0.f, 0.f, 0.f, 0.f, 0.f, 0.f, 0.f, 0.f};

  __syncthreads();

  for (int i = 0; i < NK; ++i) {
    const bf16_t* kbaseh = Km + ((size_t)(b * SEQ)) * NPROJ + i * DIM;   // B^T rows = t
    const bf16_t* vbaseh = Vt + ((size_t)(b * NK + i)) * DIM * SEQ;      // B^T rows = d

    for (int t0 = 0; t0 <= smax; t0 += 128) {
      const int tsub = t0 + wave * 16;              // wave-uniform
      if (tsub <= smax) {
        v8f sac = (v8f)(0.0f);
#pragma unroll
        for (int kk = 0; kk < 16; ++kk) {
          v16bf kf = load_b_frag(kbaseh, NPROJ, tsub + nlane, half, kk * 32);
          sac = wmma_bf16(qf[kk], kf, sac);
        }
        // causal mask + relu; stage P in LDS; accumulate denominator
#pragma unroll
        for (int r = 0; r < 8; ++r) {
          const int srow = s0 + r + half * 8;
          const int tcol = tsub + nlane;
          float v = sac[r];
          v = (tcol <= srow && v > 0.0f) ? v : 0.0f;
          dsum[r] += v;
          P[(r + half * 8) * 128 + wave * 16 + nlane] = (bf16_t)v;
        }
      } else {
#pragma unroll
        for (int r = 0; r < 8; ++r)
          P[(r + half * 8) * 128 + wave * 16 + nlane] = (bf16_t)0.0f;
      }
      __syncthreads();

      // acc(16 x 64-slice) += P(16x128) @ V(128 x 64-slice)
#pragma unroll
      for (int kbb = 0; kbb < 4; ++kbb) {
        v16bf pf = load_a_frag(P, 128, nlane, half, kbb * 32);
#pragma unroll
        for (int j = 0; j < 4; ++j) {
          const int dcol = wave * 64 + j * 16 + nlane;
          v16bf vf = load_b_frag(vbaseh, SEQ, dcol, half, t0 + kbb * 32);
          acc[j] = wmma_bf16(pf, vf, acc[j]);
        }
      }
      __syncthreads();
    }
  }

  // denominator: reduce per-row partials across the 16-lane half, combine in LDS
#pragma unroll
  for (int r = 0; r < 8; ++r) {
    float v = dsum[r];
    v += __shfl_xor(v, 1, 32);
    v += __shfl_xor(v, 2, 32);
    v += __shfl_xor(v, 4, 32);
    v += __shfl_xor(v, 8, 32);
    if (nlane == 0) atomicAdd(&denom[r + half * 8], v);
  }
  __syncthreads();

  // normalize and store (bs, seq, NQ*dim) bf16 for the output projection
#pragma unroll
  for (int j = 0; j < 4; ++j) {
    const int n = h * DIM + wave * 64 + j * 16 + nlane;
#pragma unroll
    for (int r = 0; r < 8; ++r) {
      const int m = b * SEQ + s0 + r + half * 8;
      const float inv = 1.0f / (denom[r + half * 8] + 1e-12f);
      Amid[(size_t)m * NPROJ + n] = (bf16_t)(acc[j][r] * inv);
    }
  }
}

// ---------------------------------------------------------------------------
extern "C" void kernel_launch(void* const* d_in, const int* in_sizes, int n_in,
                              void* d_out, int out_size, void* d_ws, size_t ws_size,
                              hipStream_t stream) {
  const float* x    = (const float*)d_in[0];
  const float* WK_w = (const float*)d_in[1];
  const float* WK_b = (const float*)d_in[2];
  const float* WV_w = (const float*)d_in[3];
  const float* WV_b = (const float*)d_in[4];
  const float* WQ_w = (const float*)d_in[5];
  const float* WQ_b = (const float*)d_in[6];
  const float* WO_w = (const float*)d_in[7];
  const float* WO_b = (const float*)d_in[8];

  char* ws = (char*)d_ws;
  auto carve = [&](size_t bytes) {
    char* p = ws;
    ws += (bytes + 255) & ~size_t(255);
    return p;
  };
  bf16_t* xt   = (bf16_t*)carve((size_t)MROWS * DIM * 2);    // 2 MB
  bf16_t* wk   = (bf16_t*)carve((size_t)NPROJ * DIM * 2);    // 2 MB
  bf16_t* wv   = (bf16_t*)carve((size_t)NPROJ * DIM * 2);    // 2 MB
  bf16_t* wq   = (bf16_t*)carve((size_t)NPROJ * DIM * 2);    // 2 MB
  bf16_t* wo   = (bf16_t*)carve((size_t)DIM * NPROJ * 2);    // 2 MB
  bf16_t* Km   = (bf16_t*)carve((size_t)MROWS * NPROJ * 2);  // 8 MB
  bf16_t* Qm   = (bf16_t*)carve((size_t)MROWS * NPROJ * 2);  // 8 MB
  bf16_t* Vtb  = (bf16_t*)carve((size_t)MROWS * NPROJ * 2);  // 8 MB
  bf16_t* Amid = (bf16_t*)carve((size_t)MROWS * NPROJ * 2);  // 8 MB

  const int nElemX = SEQ * BSZ * DIM;      // 1,048,576
  const int nElemW = NPROJ * DIM;          // 1,048,576
  cvt_x_kernel<<<(nElemX + 255) / 256, 256, 0, stream>>>(x, xt, nElemX);
  cvt_f32_bf16<<<(nElemW + 255) / 256, 256, 0, stream>>>(WK_w, wk, nElemW);
  cvt_f32_bf16<<<(nElemW + 255) / 256, 256, 0, stream>>>(WV_w, wv, nElemW);
  cvt_f32_bf16<<<(nElemW + 255) / 256, 256, 0, stream>>>(WQ_w, wq, nElemW);
  cvt_f32_bf16<<<(nElemW + 255) / 256, 256, 0, stream>>>(WO_w, wo, nElemW);

  const float qscale = 0.044194173824159216f;  // 1/sqrt(512)

  dim3 gProj(MROWS / 16, NPROJ / 512);  // (128, 4)
  gemm_bf16_kernel<<<gProj, 256, 0, stream>>>(xt, wk, WK_b, (void*)Km,
                                              MROWS, NPROJ, DIM, 1.0f, 0);
  gemm_bf16_kernel<<<gProj, 256, 0, stream>>>(xt, wv, WV_b, (void*)Vtb,
                                              MROWS, NPROJ, DIM, 1.0f, 1);
  gemm_bf16_kernel<<<gProj, 256, 0, stream>>>(xt, wq, WQ_b, (void*)Qm,
                                              MROWS, NPROJ, DIM, qscale, 0);

  dim3 gAttn(SEQ / 16, NQ, BSZ);        // (64, 4, 2)
  attn_kernel<<<gAttn, 256, 0, stream>>>(Qm, Km, Vtb, Amid);

  dim3 gOut(MROWS / 16, 1);             // N = 512 -> one y-slab
  gemm_bf16_kernel<<<gOut, 256, 0, stream>>>(Amid, wo, WO_b, d_out,
                                             MROWS, DIM, NPROJ, 1.0f, 2);
}
